// OmniAttention_23115513987297
// MI455X (gfx1250) — matrix operand order repaired and checked by reading, not compiled, not run on hardware
//
#include <hip/hip_runtime.h>
#include <hip/hip_bf16.h>

typedef __attribute__((ext_vector_type(16))) _Float16 v16h;
typedef __attribute__((ext_vector_type(8)))  _Float16 v8h;
typedef __attribute__((ext_vector_type(8)))  float    v8f;

#define BATCH   2
#define NSEQ    2048
#define DMODEL  1024
#define NHEAD   16
#define HDIM    64
#define SCALE_Q 0.125f   // 64^-0.5, exact power of two

// ---------------------------------------------------------------------------
// WMMA helpers (CDNA5 layouts, wave32)
// ---------------------------------------------------------------------------
__device__ __forceinline__ v8f wmma_f16(v16h a, v16h b, v8f c) {
  // D = A(16x32 f16) * B(32x16 f16) + C(16x16 f32)
  return __builtin_amdgcn_wmma_f32_16x16x32_f16(
      /*neg_a=*/false, a, /*neg_b=*/false, b,
      /*c_mod=*/(short)0, c, /*reuse_a=*/false, /*reuse_b=*/false);
}

// A-matrix fragment (16x32 f16, row-major source, row = caller-provided incl lane%16)
// lanes 0-15: elems 0-15 -> K=k+0..15
// lanes 16-31: elems 0-15 -> K=k+16..31
__device__ __forceinline__ v16h load_a_frag(const _Float16* A, int lda, int row,
                                            int k, int hi) {
  const _Float16* p = A + (size_t)row * lda + k;
  v8h lo = *(const v8h*)(p + hi * 16);
  v8h hp = *(const v8h*)(p + hi * 16 + 8);
  return __builtin_shufflevector(lo, hp, 0,1,2,3,4,5,6,7,8,9,10,11,12,13,14,15);
}

// B-matrix fragment (32x16 f16) from a transposed (col-major-of-B == [N,K]) source:
// lane owns column `col`, 16 contiguous K values starting at k + (lane/16)*16
__device__ __forceinline__ v16h load_b_frag(const _Float16* BT, int ldb, int col,
                                            int k, int hi) {
  return *(const v16h*)(BT + (size_t)col * ldb + k + hi * 16);
}

// ---------------------------------------------------------------------------
// fp32 -> f16 conversion / transpose staging
// ---------------------------------------------------------------------------
__global__ void cvt_kernel(const float* __restrict__ src, _Float16* __restrict__ dst,
                           int n) {
  int i = blockIdx.x * blockDim.x + threadIdx.x;
  if (i < n) dst[i] = (_Float16)src[i];
}

// dst[j*K + kk] = (f16) src[kk*Ncol + j]   (j < Ncol, kk < K)
__global__ void transpose_cvt_kernel(const float* __restrict__ src,
                                     _Float16* __restrict__ dst, int K, int Ncol) {
  int i = blockIdx.x * blockDim.x + threadIdx.x;
  if (i >= K * Ncol) return;
  int j = i / K, kk = i - j * K;
  dst[i] = (_Float16)src[(size_t)kk * Ncol + j];
}

// ---------------------------------------------------------------------------
// QKV GEMM: [4096,1024] x [1024,3072] -> scattered into Q,K ([B,H,N,64]) and
// V transposed ([B,H,64,N]), f16. Q pre-scaled by SCALE_Q. 32x64 wave tile.
// ---------------------------------------------------------------------------
__global__ void __launch_bounds__(128)
gemm_qkv_kernel(const _Float16* __restrict__ xh, const _Float16* __restrict__ wqkvT,
                const float* __restrict__ bqkv, _Float16* __restrict__ Qb,
                _Float16* __restrict__ Kb, _Float16* __restrict__ Vt) {
  const int lane = threadIdx.x & 31, lo16 = lane & 15, hi = lane >> 4;
  const int wave = blockIdx.x * 4 + (threadIdx.x >> 5);
  const int wn = wave % 48, wm = wave / 48;        // 48 n-tiles of 64, 128 m-tiles of 32
  const int m0 = wm * 32, n0 = wn * 64;

  v8f acc[2][4] = {};
  for (int k = 0; k < DMODEL; k += 32) {
    __builtin_prefetch(xh + (size_t)(m0 + lo16) * DMODEL + k + 64, 0, 1);
    v16h a0 = load_a_frag(xh, DMODEL, m0 + lo16, k, hi);
    v16h a1 = load_a_frag(xh, DMODEL, m0 + 16 + lo16, k, hi);
#pragma unroll
    for (int j = 0; j < 4; ++j) {
      v16h bf = load_b_frag(wqkvT, DMODEL, n0 + j * 16 + lo16, k, hi);
      acc[0][j] = wmma_f16(a0, bf, acc[0][j]);
      acc[1][j] = wmma_f16(a1, bf, acc[1][j]);
    }
  }

#pragma unroll
  for (int i = 0; i < 2; ++i)
#pragma unroll
    for (int j = 0; j < 4; ++j) {
      const int col = n0 + j * 16 + lo16;       // [0,3072)
      const int which = col >> 10;              // 0=Q 1=K 2=V (uniform per tile)
      const int d = col & 1023;
      const int h = d >> 6, hd = d & 63;
      const float bias = bqkv[col];
#pragma unroll
      for (int r = 0; r < 8; ++r) {
        const int row = m0 + i * 16 + r + 8 * hi;   // token in [0,4096)
        const int bb = row >> 11, nn = row & (NSEQ - 1);
        const int bh = bb * NHEAD + h;
        const float v = acc[i][j][r] + bias;
        if (which == 0)
          Qb[((size_t)bh * NSEQ + nn) * HDIM + hd] = (_Float16)(v * SCALE_Q);
        else if (which == 1)
          Kb[((size_t)bh * NSEQ + nn) * HDIM + hd] = (_Float16)v;
        else
          Vt[((size_t)bh * HDIM + hd) * NSEQ + nn] = (_Float16)v;
      }
    }
}

// ---------------------------------------------------------------------------
// Flash attention: one wave per (b, h, 16-query tile); stream 32 keys/iter.
// S via 4 WMMAs, online softmax (shfl within 16-lane halves), P staged through
// per-wave LDS tile to become an A-fragment, P*V via 4 WMMAs.
// ---------------------------------------------------------------------------
__global__ void __launch_bounds__(128)
attention_kernel(const _Float16* __restrict__ Qb, const _Float16* __restrict__ Kb,
                 const _Float16* __restrict__ Vt, const float* __restrict__ mask,
                 _Float16* __restrict__ ctx) {
  __shared__ __align__(32) _Float16 pbuf[4][16 * 32];
  const int lane = threadIdx.x & 31, lo16 = lane & 15, hi = lane >> 4;
  const int wid = threadIdx.x >> 5;
  const int task = blockIdx.x * 4 + wid;        // 4096 = 2*16*128 tasks
  const int b = task >> 11;
  const int rem = task & 2047;
  const int h = rem >> 7;
  const int q0 = (rem & 127) << 4;

  const _Float16* Qh = Qb + ((size_t)(b * NHEAD + h)) * NSEQ * HDIM;
  const _Float16* Kh = Kb + ((size_t)(b * NHEAD + h)) * NSEQ * HDIM;
  const _Float16* Vh = Vt + ((size_t)(b * NHEAD + h)) * HDIM * NSEQ;
  const float* mrow = mask + (size_t)b * NSEQ * NSEQ;

  const v16h qa0 = load_a_frag(Qh, HDIM, q0 + lo16, 0, hi);
  const v16h qa1 = load_a_frag(Qh, HDIM, q0 + lo16, 32, hi);

  v8f o[4] = {};
  v8f mrun, lrun;
#pragma unroll
  for (int r = 0; r < 8; ++r) { mrun[r] = -1e30f; lrun[r] = 0.0f; }

  _Float16* pw = pbuf[wid];

  for (int k0 = 0; k0 < NSEQ; k0 += 32) {
    // --- S = (Q*scale) @ K^T + mask, two 16x16 tiles covering 32 keys ---
    v8f s0 = {}, s1 = {};
    {
      v16h kb0 = load_b_frag(Kh, HDIM, k0 + lo16, 0, hi);
      v16h kb1 = load_b_frag(Kh, HDIM, k0 + lo16, 32, hi);
      v16h kb2 = load_b_frag(Kh, HDIM, k0 + 16 + lo16, 0, hi);
      v16h kb3 = load_b_frag(Kh, HDIM, k0 + 16 + lo16, 32, hi);
      s0 = wmma_f16(qa0, kb0, s0);
      s0 = wmma_f16(qa1, kb1, s0);
      s1 = wmma_f16(qa0, kb2, s1);
      s1 = wmma_f16(qa1, kb3, s1);
    }
#pragma unroll
    for (int r = 0; r < 8; ++r) {
      const int q = q0 + r + 8 * hi;
      const float* mp = mrow + (size_t)q * NSEQ + k0;
      s0[r] += mp[lo16];
      s1[r] += mp[16 + lo16];
    }

    // --- online softmax update (rows live across 16-lane halves) ---
    v8f mnew, alpha;
#pragma unroll
    for (int r = 0; r < 8; ++r) {
      float t = fmaxf(s0[r], s1[r]);
      t = fmaxf(t, __shfl_xor(t, 8, 16));
      t = fmaxf(t, __shfl_xor(t, 4, 16));
      t = fmaxf(t, __shfl_xor(t, 2, 16));
      t = fmaxf(t, __shfl_xor(t, 1, 16));
      const float mn = fmaxf(mrun[r], t);
      mnew[r] = mn;
      alpha[r] = __expf(mrun[r] - mn);
      mrun[r] = mn;
    }
#pragma unroll
    for (int r = 0; r < 8; ++r) {
      s0[r] = __expf(s0[r] - mnew[r]);
      s1[r] = __expf(s1[r] - mnew[r]);
      float t = s0[r] + s1[r];
      t += __shfl_xor(t, 8, 16);
      t += __shfl_xor(t, 4, 16);
      t += __shfl_xor(t, 2, 16);
      t += __shfl_xor(t, 1, 16);
      lrun[r] = lrun[r] * alpha[r] + t;
    }
#pragma unroll
    for (int t4 = 0; t4 < 4; ++t4)
#pragma unroll
      for (int r = 0; r < 8; ++r) o[t4][r] *= alpha[r];

    // --- C-layout P -> LDS -> A-layout fragment (f16) ---
#pragma unroll
    for (int r = 0; r < 8; ++r) {
      const int row = r + 8 * hi;
      pw[row * 32 + lo16]      = (_Float16)s0[r];
      pw[row * 32 + 16 + lo16] = (_Float16)s1[r];
    }
    asm volatile("s_wait_dscnt 0" ::: "memory");
    v8h plo = *(const v8h*)(pw + lo16 * 32 + hi * 16);
    v8h phi = *(const v8h*)(pw + lo16 * 32 + hi * 16 + 8);
    v16h pa = __builtin_shufflevector(plo, phi,
                                      0,1,2,3,4,5,6,7,8,9,10,11,12,13,14,15);

    // --- O += P @ V  (V stored transposed: key-contiguous per hd row) ---
#pragma unroll
    for (int t4 = 0; t4 < 4; ++t4) {
      v16h vb = load_b_frag(Vh, NSEQ, t4 * 16 + lo16, k0, hi);
      o[t4] = wmma_f16(pa, vb, o[t4]);
    }
  }

  // --- normalize and scatter context as f16 [B, N, D] ---
#pragma unroll
  for (int r = 0; r < 8; ++r) {
    const float inv = 1.0f / lrun[r];
    const int q = q0 + r + 8 * hi;
    const size_t base = ((size_t)b * NSEQ + q) * DMODEL + h * HDIM;
#pragma unroll
    for (int t4 = 0; t4 < 4; ++t4)
      ctx[base + t4 * 16 + lo16] = (_Float16)(o[t4][r] * inv);
  }
}

// ---------------------------------------------------------------------------
// Output projection: [4096,1024] x [1024,1024] + bias -> fp32 out
// ---------------------------------------------------------------------------
__global__ void __launch_bounds__(128)
gemm_proj_kernel(const _Float16* __restrict__ ctxh, const _Float16* __restrict__ wprojT,
                 const float* __restrict__ bproj, float* __restrict__ out) {
  const int lane = threadIdx.x & 31, lo16 = lane & 15, hi = lane >> 4;
  const int wave = blockIdx.x * 4 + (threadIdx.x >> 5);
  const int wn = wave & 15, wm = wave >> 4;       // 16 n-tiles of 64, 128 m-tiles of 32
  const int m0 = wm * 32, n0 = wn * 64;

  v8f acc[2][4] = {};
  for (int k = 0; k < DMODEL; k += 32) {
    __builtin_prefetch(ctxh + (size_t)(m0 + lo16) * DMODEL + k + 64, 0, 1);
    v16h a0 = load_a_frag(ctxh, DMODEL, m0 + lo16, k, hi);
    v16h a1 = load_a_frag(ctxh, DMODEL, m0 + 16 + lo16, k, hi);
#pragma unroll
    for (int j = 0; j < 4; ++j) {
      v16h bf = load_b_frag(wprojT, DMODEL, n0 + j * 16 + lo16, k, hi);
      acc[0][j] = wmma_f16(a0, bf, acc[0][j]);
      acc[1][j] = wmma_f16(a1, bf, acc[1][j]);
    }
  }

#pragma unroll
  for (int i = 0; i < 2; ++i)
#pragma unroll
    for (int j = 0; j < 4; ++j) {
      const int col = n0 + j * 16 + lo16;
      const float bias = bproj[col];
#pragma unroll
      for (int r = 0; r < 8; ++r) {
        const int row = m0 + i * 16 + r + 8 * hi;
        out[(size_t)row * DMODEL + col] = acc[i][j][r] + bias;
      }
    }
}

// ---------------------------------------------------------------------------
// Launch: convert/stage -> QKV GEMM -> flash attention -> proj GEMM
// ---------------------------------------------------------------------------
extern "C" void kernel_launch(void* const* d_in, const int* in_sizes, int n_in,
                              void* d_out, int out_size, void* d_ws, size_t ws_size,
                              hipStream_t stream) {
  (void)in_sizes; (void)n_in; (void)out_size; (void)ws_size;
  const float* x     = (const float*)d_in[0];
  const float* mask  = (const float*)d_in[1];
  const float* wqkv  = (const float*)d_in[2];
  const float* bqkv  = (const float*)d_in[3];
  const float* wproj = (const float*)d_in[4];
  const float* bproj = (const float*)d_in[5];
  float* out = (float*)d_out;

  // f16 staging workspace (48 MB total)
  _Float16* xh     = (_Float16*)d_ws;                       // 4096*1024
  _Float16* wqkvT  = xh     + (size_t)4096 * 1024;          // 3072*1024 [out,in]
  _Float16* wprojT = wqkvT  + (size_t)3072 * 1024;          // 1024*1024 [out,in]
  _Float16* Qb     = wprojT + (size_t)1024 * 1024;          // [B,H,N,64], pre-scaled
  _Float16* Kb     = Qb     + (size_t)4194304;              // [B,H,N,64]
  _Float16* Vt     = Kb     + (size_t)4194304;              // [B,H,64,N]
  _Float16* ctx    = Vt     + (size_t)4194304;              // [B,N,D]

  cvt_kernel<<<4194304 / 256, 256, 0, stream>>>(x, xh, 4194304);
  transpose_cvt_kernel<<<3145728 / 256, 256, 0, stream>>>(wqkv, wqkvT, 1024, 3072);
  transpose_cvt_kernel<<<1048576 / 256, 256, 0, stream>>>(wproj, wprojT, 1024, 1024);

  gemm_qkv_kernel<<<1536, 128, 0, stream>>>(xh, wqkvT, bqkv, Qb, Kb, Vt);
  attention_kernel<<<1024, 128, 0, stream>>>(Qb, Kb, Vt, mask, ctx);
  gemm_proj_kernel<<<512, 128, 0, stream>>>(ctx, wprojT, bproj, out);
}
